// VirtGATMolNode_33947421507891
// MI455X (gfx1250) — matrix operand order repaired and checked by reading, not compiled, not run on hardware
//
#include <hip/hip_runtime.h>
#include <hip/hip_bf16.h>

// ---------------- problem constants ----------------
#define NN 100000
#define EE 400000
#define ET 500000   // E + N (self loops)
#define BB 500
#define DD 256
#define HH 4
#define CC 64
#define AV 64
#define BV 8
#define LL 3
#define EPSV 1e-5f
#define SLOPE 0.2f

typedef __attribute__((ext_vector_type(16))) __bf16 v16bf;
typedef __attribute__((ext_vector_type(8)))  __bf16 v8bf;
typedef __attribute__((ext_vector_type(8)))  float  v8f;

// ---------------- small device helpers ----------------
__device__ __forceinline__ unsigned short f2bf(float f) {
  unsigned u = __float_as_uint(f);
  unsigned r = u + 0x7FFFu + ((u >> 16) & 1u);   // round-to-nearest-even
  return (unsigned short)(r >> 16);
}
__device__ __forceinline__ unsigned enc_f(float x) {
  unsigned u = __float_as_uint(x);
  return (u & 0x80000000u) ? ~u : (u | 0x80000000u);
}
__device__ __forceinline__ float dec_f(unsigned k) {
  return (k & 0x80000000u) ? __uint_as_float(k & 0x7FFFFFFFu) : __uint_as_float(~k);
}

// ---------------- generic utility kernels ----------------
__global__ void k_zero_f(float* p, int n) {
  int i = blockIdx.x * blockDim.x + threadIdx.x;
  if (i < n) p[i] = 0.f;
}
__global__ void k_zero_i(int* p, int n) {
  int i = blockIdx.x * blockDim.x + threadIdx.x;
  if (i < n) p[i] = 0;
}
__global__ void k_set_u32(unsigned* p, unsigned v, int n) {
  int i = blockIdx.x * blockDim.x + threadIdx.x;
  if (i < n) p[i] = v;
}
__global__ void k_copy_f(const float* __restrict__ a, float* __restrict__ b, int n) {
  int i = blockIdx.x * blockDim.x + threadIdx.x;
  if (i < n) b[i] = a[i];
}

// ---------------- self-loop fill = per-dst mean of raw bond int features ----------------
__global__ void k_deg_esum(const int* __restrict__ ei, const int* __restrict__ ea,
                           int* __restrict__ deg, int* __restrict__ esum) {
  int e = blockIdx.x * blockDim.x + threadIdx.x;
  if (e >= EE) return;
  int d = ei[EE + e];
  atomicAdd(&deg[d], 1);
  atomicAdd(&esum[d * 3 + 0], ea[e * 3 + 0]);
  atomicAdd(&esum[d * 3 + 1], ea[e * 3 + 1]);
  atomicAdd(&esum[d * 3 + 2], ea[e * 3 + 2]);
}
__global__ void k_fillmean(const int* __restrict__ deg, const int* __restrict__ esum,
                           int* __restrict__ fill) {
  int i = blockIdx.x * blockDim.x + threadIdx.x;
  if (i >= NN * 3) return;
  int dg = deg[i / 3];
  fill[i] = esum[i] / (dg > 0 ? dg : 1);   // floor-div on non-negative ints
}

// ---------------- atom encode ----------------
__global__ void k_atom_encode(const int* __restrict__ x, const float* __restrict__ emb,
                              float* __restrict__ h) {
  int idx = blockIdx.x * blockDim.x + threadIdx.x;
  if (idx >= NN * DD) return;
  int n = idx >> 8, d = idx & 255;
  float acc = 0.f;
  #pragma unroll
  for (int f = 0; f < 9; ++f)
    acc += emb[((size_t)f * AV + x[n * 9 + f]) * DD + d];
  h[idx] = acc;
}

// ---------------- virtual-node broadcast / add / pool ----------------
__global__ void k_broadcast_row(const float* __restrict__ row, float* __restrict__ out, int n) {
  int i = blockIdx.x * blockDim.x + threadIdx.x;
  if (i < n) out[i] = row[i & 255];
}
__global__ void k_add_virt(float* __restrict__ h, const float* __restrict__ virt,
                           const int* __restrict__ batch) {
  int idx = blockIdx.x * blockDim.x + threadIdx.x;
  if (idx >= NN * DD) return;
  int n = idx >> 8, d = idx & 255;
  h[idx] += virt[batch[n] * DD + d];
}
__global__ void k_pool(const float* __restrict__ h, const int* __restrict__ batch,
                       float* __restrict__ vt) {
  int idx = blockIdx.x * blockDim.x + threadIdx.x;
  if (idx >= NN * DD) return;
  int n = idx >> 8, d = idx & 255;
  atomicAdd(&vt[batch[n] * DD + d], h[idx]);
}

// ---------------- bf16 staging + weight packing for WMMA ----------------
__global__ void k_f2bf(const float* __restrict__ in, unsigned short* __restrict__ out, int n) {
  int i = blockIdx.x * blockDim.x + threadIdx.x;
  if (i < n) out[i] = f2bf(in[i]);
}
// Pack W[k][n] (row-major 256x256) into per-lane contiguous B-fragments:
// P[((jt*8 + k)*32 + lane)*16 + i] = bf16(W[(k*32 + (i<8 ? (lane>>4)*8 + i : 16 + (lane>>4)*8 + i-8))*256 + jt*16 + (lane&15)])
__global__ void k_pack_w(const float* __restrict__ W, unsigned short* __restrict__ P) {
  int idx = blockIdx.x * blockDim.x + threadIdx.x;
  if (idx >= 65536) return;
  int i    = idx & 15;
  int lane = (idx >> 4) & 31;
  int k    = (idx >> 9) & 7;
  int jt   = idx >> 12;
  int half = lane >> 4, lrow = lane & 15;
  P[idx] = f2bf(W[(size_t)(k * 32 + (i < 8 ? half * 8 + i : 16 + half * 8 + (i - 8))) * DD + jt * 16 + lrow]);
}

// ---------------- WMMA GEMM: xs[100000x256] = hbf[100000x256] @ Wpack ----------------
// Register-blocked 1x4: one wave computes a 16x64 output strip. One A fragment is
// shared by 4 B fragments / 4 accumulators per K-step -> 32 v_wmma per wave,
// 2.5 b128 loads per WMMA, A traffic cut 4x vs one-tile-per-wave.
__global__ __launch_bounds__(128) void k_wmma_gemm(const unsigned short* __restrict__ A,
                                                   const unsigned short* __restrict__ Bp,
                                                   float* __restrict__ Cmat) {
  int wave = threadIdx.x >> 5;         // 0..3
  int lane = threadIdx.x & 31;
  int half = lane >> 4;
  int lrow = lane & 15;
  int mt = blockIdx.x;                 // 0..6249 (100000/16 exactly)
  int jt0 = wave * 4;                  // column tiles jt0..jt0+3
  const unsigned short* arow = A + (size_t)(mt * 16 + lrow) * DD;
  const v16bf* bp = (const v16bf*)Bp;
  v8f c0 = {}, c1 = {}, c2 = {}, c3 = {};
  #pragma unroll
  for (int k = 0; k < 8; ++k) {
    int kbase = k * 32;
    // 16-bit A 16x32 layout: lane<16 holds K {0..7,16..23}, lane>=16 holds {8..15,24..31}
    v8bf lo = *(const v8bf*)(arow + kbase + half * 8);
    v8bf hi = *(const v8bf*)(arow + kbase + 16 + half * 8);
    v16bf a = __builtin_shufflevector(lo, hi, 0,1,2,3,4,5,6,7,8,9,10,11,12,13,14,15);
    v16bf b0 = bp[((jt0 + 0) * 8 + k) * 32 + lane];
    v16bf b1 = bp[((jt0 + 1) * 8 + k) * 32 + lane];
    v16bf b2 = bp[((jt0 + 2) * 8 + k) * 32 + lane];
    v16bf b3 = bp[((jt0 + 3) * 8 + k) * 32 + lane];
    c0 = __builtin_amdgcn_wmma_f32_16x16x32_bf16(false, a, false, b0, (short)0, c0, false, false);
    c1 = __builtin_amdgcn_wmma_f32_16x16x32_bf16(false, a, false, b1, (short)0, c1, false, false);
    c2 = __builtin_amdgcn_wmma_f32_16x16x32_bf16(false, a, false, b2, (short)0, c2, false, false);
    c3 = __builtin_amdgcn_wmma_f32_16x16x32_bf16(false, a, false, b3, (short)0, c3, false, false);
  }
  // C layout: VGPR r -> row r (lanes 0-15) / r+8 (lanes 16-31), col = lane&15
  float* crow = Cmat + (size_t)(mt * 16 + half * 8) * DD + jt0 * 16 + lrow;
  #pragma unroll
  for (int r = 0; r < 8; ++r) {
    crow[(size_t)r * DD + 0]  = c0[r];
    crow[(size_t)r * DD + 16] = c1[r];
    crow[(size_t)r * DD + 32] = c2[r];
    crow[(size_t)r * DD + 48] = c3[r];
  }
}

// ---------------- attention scores per (node, head) ----------------
__global__ void k_attn_scores(const float* __restrict__ xs, const float* __restrict__ asrc,
                              const float* __restrict__ adst,
                              float* __restrict__ as_out, float* __restrict__ ad_out) {
  int idx = blockIdx.x * blockDim.x + threadIdx.x;
  if (idx >= NN * HH) return;
  int n = idx >> 2, hh = idx & 3;
  const float* xrow = xs + (size_t)n * DD + hh * CC;
  const float* ws = asrc + hh * CC;
  const float* wd = adst + hh * CC;
  float s1 = 0.f, s2 = 0.f;
  #pragma unroll 4
  for (int ci = 0; ci < CC; ++ci) { float v = xrow[ci]; s1 += v * ws[ci]; s2 += v * wd[ci]; }
  as_out[idx] = s1;
  ad_out[idx] = s2;
}

// ---------------- edge pass 1: leaky-relu logits + segment max ----------------
__global__ void k_alpha1(const int* __restrict__ ei, const int* __restrict__ ea,
                         const int* __restrict__ fill, const float* __restrict__ be,
                         const float* __restrict__ atte, const float* __restrict__ as_in,
                         const float* __restrict__ ad_in, float* __restrict__ alpha,
                         unsigned* __restrict__ amaxk) {
  int idx = blockIdx.x * blockDim.x + threadIdx.x;
  if (idx >= ET * HH) return;
  int e = idx >> 2, hh = idx & 3;
  int s, d, a0, a1, a2;
  if (e < EE) { s = ei[e]; d = ei[EE + e]; a0 = ea[e*3]; a1 = ea[e*3+1]; a2 = ea[e*3+2]; }
  else { int n = e - EE; s = n; d = n; a0 = fill[n*3]; a1 = fill[n*3+1]; a2 = fill[n*3+2]; }
  const float* p0 = be + ((size_t)(0 * BV + a0)) * DD + hh * CC;
  const float* p1 = be + ((size_t)(1 * BV + a1)) * DD + hh * CC;
  const float* p2 = be + ((size_t)(2 * BV + a2)) * DD + hh * CC;
  const float* w  = atte + hh * CC;
  float dot = 0.f;
  #pragma unroll 4
  for (int ci = 0; ci < CC; ++ci) dot += (p0[ci] + p1[ci] + p2[ci]) * w[ci];
  float a = ad_in[d * HH + hh] + as_in[s * HH + hh] + dot;
  a = a > 0.f ? a : a * SLOPE;
  alpha[idx] = a;
  atomicMax(&amaxk[d * HH + hh], enc_f(a));
}

// ---------------- edge pass 2: exp(a - max) + segment sum ----------------
__global__ void k_alpha2(const int* __restrict__ ei, float* __restrict__ alpha,
                         const unsigned* __restrict__ amaxk, float* __restrict__ denom) {
  int idx = blockIdx.x * blockDim.x + threadIdx.x;
  if (idx >= ET * HH) return;
  int e = idx >> 2, hh = idx & 3;
  int d = (e < EE) ? ei[EE + e] : (e - EE);
  float ex = __expf(alpha[idx] - dec_f(amaxk[d * HH + hh]));
  alpha[idx] = ex;
  atomicAdd(&denom[d * HH + hh], ex);
}

// ---------------- edge pass 3: normalize ----------------
__global__ void k_alpha3(const int* __restrict__ ei, float* __restrict__ alpha,
                         const float* __restrict__ denom) {
  int idx = blockIdx.x * blockDim.x + threadIdx.x;
  if (idx >= ET * HH) return;
  int e = idx >> 2, hh = idx & 3;
  int d = (e < EE) ? ei[EE + e] : (e - EE);
  alpha[idx] /= denom[d * HH + hh];
}

// ---------------- out init (conv bias broadcast) ----------------
__global__ void k_init_out(float* __restrict__ out, const float* __restrict__ bias) {
  int idx = blockIdx.x * blockDim.x + threadIdx.x;
  if (idx >= NN * DD) return;
  out[idx] = bias[idx & 255];
}

// ---------------- message scatter: out[dst] += alpha * (xs[src] + e) ----------------
__global__ void k_message(const int* __restrict__ ei, const int* __restrict__ ea,
                          const int* __restrict__ fill, const float* __restrict__ be,
                          const float* __restrict__ xs, const float* __restrict__ alpha,
                          float* __restrict__ out) {
  int idx = blockIdx.x * blockDim.x + threadIdx.x;
  if (idx >= ET * DD) return;
  int e = idx >> 8, d = idx & 255, hh = d >> 6;
  int s, dn, a0, a1, a2;
  if (e < EE) { s = ei[e]; dn = ei[EE + e]; a0 = ea[e*3]; a1 = ea[e*3+1]; a2 = ea[e*3+2]; }
  else { int n = e - EE; s = n; dn = n; a0 = fill[n*3]; a1 = fill[n*3+1]; a2 = fill[n*3+2]; }
  float ec = be[((size_t)(0 * BV + a0)) * DD + d]
           + be[((size_t)(1 * BV + a1)) * DD + d]
           + be[((size_t)(2 * BV + a2)) * DD + d];
  float coef = alpha[e * HH + hh];
  float val = coef * (xs[(size_t)s * DD + d] + ec);
  atomicAdd(&out[(size_t)dn * DD + d], val);
}

// ---------------- column stats (partial; blockDim.x == cols) ----------------
__global__ void k_colstat(const float* __restrict__ X, int rows, int cols, int rowsPerBlock,
                          float* __restrict__ s, float* __restrict__ sq) {
  int c = threadIdx.x;
  int r0 = blockIdx.x * rowsPerBlock;
  int r1 = r0 + rowsPerBlock; if (r1 > rows) r1 = rows;
  float sum = 0.f, ssq = 0.f;
  for (int r = r0; r < r1; ++r) {
    float v = X[(size_t)r * cols + c];
    sum += v; ssq += v * v;
  }
  atomicAdd(&s[c], sum);
  atomicAdd(&sq[c], ssq);
}

// ---------------- BN apply (+optional ReLU) ----------------
__global__ void k_bn_apply(const float* __restrict__ X, float* __restrict__ Y,
                           const float* __restrict__ s, const float* __restrict__ sq,
                           const float* __restrict__ g, const float* __restrict__ beta,
                           int total, int cols, float rinv, int relu) {
  int idx = blockIdx.x * blockDim.x + threadIdx.x;
  if (idx >= total) return;
  int c = idx % cols;
  float m = s[c] * rinv;
  float v = sq[c] * rinv - m * m;
  float y = (X[idx] - m) * rsqrtf(v + EPSV) * g[c] + beta[c];
  if (relu) y = fmaxf(y, 0.f);
  Y[idx] = y;
}

// ---------------- small dense GEMM (virtual-node MLPs) ----------------
__global__ void k_gemm_small(const float* __restrict__ A, const float* __restrict__ Wt,
                             const float* __restrict__ bias, float* __restrict__ out,
                             int M, int K, int Nc) {
  int idx = blockIdx.x * blockDim.x + threadIdx.x;
  if (idx >= M * Nc) return;
  int m = idx / Nc, n = idx % Nc;
  float acc = bias[n];
  const float* a = A + (size_t)m * K;
  for (int k = 0; k < K; ++k) acc += a[k] * Wt[(size_t)k * Nc + n];
  out[idx] = acc;
}

// =====================================================================
extern "C" void kernel_launch(void* const* d_in, const int* in_sizes, int n_in,
                              void* d_out, int out_size, void* d_ws, size_t ws_size,
                              hipStream_t stream) {
  const int*   x         = (const int*)d_in[0];
  const int*   edge_index= (const int*)d_in[1];
  const int*   edge_attr = (const int*)d_in[2];
  const int*   batch     = (const int*)d_in[3];
  const float* atom_emb  = (const float*)d_in[4];
  const float* bond_emb  = (const float*)d_in[5];
  const float* Wmat      = (const float*)d_in[6];
  const float* att_src   = (const float*)d_in[7];
  const float* att_dst   = (const float*)d_in[8];
  const float* att_edge  = (const float*)d_in[9];
  const float* conv_bias = (const float*)d_in[10];
  const float* bn_g      = (const float*)d_in[11];
  const float* bn_b      = (const float*)d_in[12];
  const float* virt_emb0 = (const float*)d_in[13];
  const float* vm_w1     = (const float*)d_in[14];
  const float* vm_b1     = (const float*)d_in[15];
  const float* vm_g1     = (const float*)d_in[16];
  const float* vm_beta1  = (const float*)d_in[17];
  const float* vm_w2     = (const float*)d_in[18];
  const float* vm_b2     = (const float*)d_in[19];
  const float* vm_g2     = (const float*)d_in[20];
  const float* vm_beta2  = (const float*)d_in[21];

  // ---- workspace layout ----
  char* base = (char*)d_ws;
  size_t off = 0;
  auto alloc = [&](size_t bytes) { char* p = base + off; off += (bytes + 255) & ~(size_t)255; return p; };
  float*          h0     = (float*)alloc((size_t)NN * DD * 4);
  float*          h1     = (float*)alloc((size_t)NN * DD * 4);
  float*          xs     = (float*)alloc((size_t)NN * DD * 4);
  unsigned short* hbf    = (unsigned short*)alloc((size_t)NN * DD * 2);
  unsigned short* wpack  = (unsigned short*)alloc((size_t)DD * DD * 2);
  float*          asb    = (float*)alloc((size_t)NN * HH * 4);
  float*          adb    = (float*)alloc((size_t)NN * HH * 4);
  float*          alpha  = (float*)alloc((size_t)ET * HH * 4);
  unsigned*       amaxk  = (unsigned*)alloc((size_t)NN * HH * 4);
  float*          denom  = (float*)alloc((size_t)NN * HH * 4);
  int*            deg    = (int*)alloc((size_t)NN * 4);
  int*            esum   = (int*)alloc((size_t)NN * 3 * 4);
  int*            fill   = (int*)alloc((size_t)NN * 3 * 4);
  float*          virt   = (float*)alloc((size_t)BB * DD * 4);
  float*          vt     = (float*)alloc((size_t)BB * DD * 4);
  float*          z      = (float*)alloc((size_t)BB * 2 * DD * 4);
  float*          vpre   = (float*)alloc((size_t)BB * DD * 4);
  float*          ssum   = (float*)alloc(512 * 4);
  float*          ssq    = (float*)alloc(512 * 4);
  (void)ws_size; (void)in_sizes; (void)n_in; (void)out_size;

  const int T = 256;
  auto GB = [&](int n) { return (n + T - 1) / T; };

  // ---- self-loop fill values ----
  k_zero_i<<<GB(NN), T, 0, stream>>>(deg, NN);
  k_zero_i<<<GB(NN * 3), T, 0, stream>>>(esum, NN * 3);
  k_deg_esum<<<GB(EE), T, 0, stream>>>(edge_index, edge_attr, deg, esum);
  k_fillmean<<<GB(NN * 3), T, 0, stream>>>(deg, esum, fill);

  // ---- atom encode + virtual node init ----
  k_atom_encode<<<GB(NN * DD), T, 0, stream>>>(x, atom_emb, h0);
  k_broadcast_row<<<GB(BB * DD), T, 0, stream>>>(virt_emb0, virt, BB * DD);

  float* hA = h0;
  float* hB = h1;
  for (int l = 0; l < LL; ++l) {
    // h = h + virt[batch]
    k_add_virt<<<GB(NN * DD), T, 0, stream>>>(hA, virt, batch);

    // xs = h @ W[l]  (bf16 WMMA, f32 accumulate, 1x4 register blocking)
    k_f2bf<<<GB(NN * DD), T, 0, stream>>>(hA, hbf, NN * DD);
    k_pack_w<<<GB(65536), T, 0, stream>>>(Wmat + (size_t)l * DD * DD, wpack);
    k_wmma_gemm<<<dim3(NN / 16), 128, 0, stream>>>(hbf, wpack, xs);

    // per-(node,head) attention scores
    k_attn_scores<<<GB(NN * HH), T, 0, stream>>>(xs, att_src + l * HH * CC,
                                                 att_dst + l * HH * CC, asb, adb);

    // segment softmax over incoming edges
    k_set_u32<<<GB(NN * HH), T, 0, stream>>>(amaxk, 0x007FFFFFu /* enc(-inf) */, NN * HH);
    k_zero_f<<<GB(NN * HH), T, 0, stream>>>(denom, NN * HH);
    const float* be = bond_emb + (size_t)l * 3 * BV * DD;
    k_alpha1<<<GB(ET * HH), T, 0, stream>>>(edge_index, edge_attr, fill, be,
                                            att_edge + l * HH * CC, asb, adb, alpha, amaxk);
    k_alpha2<<<GB(ET * HH), T, 0, stream>>>(edge_index, alpha, amaxk, denom);
    k_alpha3<<<GB(ET * HH), T, 0, stream>>>(edge_index, alpha, denom);

    // message aggregation
    float* out = (l == LL - 1) ? (float*)d_out : hB;
    k_init_out<<<GB(NN * DD), T, 0, stream>>>(out, conv_bias + l * DD);
    k_message<<<GB(ET * DD), T, 0, stream>>>(edge_index, edge_attr, fill, be, xs, alpha, out);

    // BatchNorm (+ReLU except last layer)
    k_zero_f<<<GB(2 * DD), T, 0, stream>>>(ssum, DD);
    k_zero_f<<<GB(2 * DD), T, 0, stream>>>(ssq, DD);
    k_colstat<<<(NN + 255) / 256, DD, 0, stream>>>(out, NN, DD, 256, ssum, ssq);
    k_bn_apply<<<GB(NN * DD), T, 0, stream>>>(out, out, ssum, ssq, bn_g + l * DD,
                                              bn_b + l * DD, NN * DD, DD,
                                              1.0f / (float)NN, (l < LL - 1) ? 1 : 0);

    if (l < LL - 1) {
      // virtual node update: vt = global_add_pool(h) + virt
      k_copy_f<<<GB(BB * DD), T, 0, stream>>>(virt, vt, BB * DD);
      k_pool<<<GB(NN * DD), T, 0, stream>>>(hA, batch, vt);
      // z = relu(bn(vt @ w1 + b1))
      k_gemm_small<<<GB(BB * 2 * DD), T, 0, stream>>>(vt, vm_w1 + (size_t)l * DD * 2 * DD,
                                                      vm_b1 + l * 2 * DD, z, BB, DD, 2 * DD);
      k_zero_f<<<GB(2 * DD), T, 0, stream>>>(ssum, 2 * DD);
      k_zero_f<<<GB(2 * DD), T, 0, stream>>>(ssq, 2 * DD);
      k_colstat<<<(BB + 63) / 64, 2 * DD, 0, stream>>>(z, BB, 2 * DD, 64, ssum, ssq);
      k_bn_apply<<<GB(BB * 2 * DD), T, 0, stream>>>(z, z, ssum, ssq, vm_g1 + l * 2 * DD,
                                                    vm_beta1 + l * 2 * DD, BB * 2 * DD,
                                                    2 * DD, 1.0f / (float)BB, 1);
      // virt = relu(bn(z @ w2 + b2))
      k_gemm_small<<<GB(BB * DD), T, 0, stream>>>(z, vm_w2 + (size_t)l * 2 * DD * DD,
                                                  vm_b2 + l * DD, vpre, BB, 2 * DD, DD);
      k_zero_f<<<GB(2 * DD), T, 0, stream>>>(ssum, DD);
      k_zero_f<<<GB(2 * DD), T, 0, stream>>>(ssq, DD);
      k_colstat<<<(BB + 63) / 64, DD, 0, stream>>>(vpre, BB, DD, 64, ssum, ssq);
      k_bn_apply<<<GB(BB * DD), T, 0, stream>>>(vpre, virt, ssum, ssq, vm_g2 + l * DD,
                                                vm_beta2 + l * DD, BB * DD, DD,
                                                1.0f / (float)BB, 1);
      // swap h buffers: next layer's input is this layer's output
      float* tmp = hA; hA = hB; hB = tmp;
    }
  }
}
